// AdaDecoder_38955353374866
// MI455X (gfx1250) — matrix operand order repaired and checked by reading, not compiled, not run on hardware
//
#include <hip/hip_runtime.h>
#include <hip/hip_bf16.h>

// ---------------------------------------------------------------------------
// Types for CDNA5 WMMA
// ---------------------------------------------------------------------------
typedef __attribute__((ext_vector_type(16))) __bf16          bf16x16;
typedef __attribute__((ext_vector_type(16))) unsigned short  u16x16;
typedef __attribute__((ext_vector_type(8)))  float           f32x8;

__device__ __forceinline__ unsigned short f2bf(float f) {
    unsigned int u = __float_as_uint(f);
    unsigned int r = u + 0x7fffu + ((u >> 16) & 1u);   // round-to-nearest-even
    return (unsigned short)(r >> 16);
}
__device__ __forceinline__ float sigf(float x) { return 1.0f / (1.0f + expf(-x)); }

// Problem constants
#define BB   32
#define ATT  256
#define CTX  512
#define HID  512
#define EMB  256
#define VOC  5000
#define TT   161
#define NSTEP 160
#define ZK   1280          // fused A panel: [xt(256) | prev_out(512) | h(512)]
#define ZKT  40            // 1280/32 k-tiles
#define GN   2560          // gates(2048) ++ n5(512)

// ---------------------------------------------------------------------------
// Packed A-fragment indexing (16-bit A 16x32 WMMA layout).
//   fragment(mtile, ktile): 32 lanes x 16 bf16, lane = g*16 + (row%16),
//   element j: k = ktile*32 + g*8 + (j<8 ? j : j+8)
// Buffer layout: [(mtile*apStride + kt0 + ktile)*32 + lane]*16 + j
// ---------------------------------------------------------------------------
__device__ __forceinline__ size_t packA_idx(int row, int k, int apStride, int kt0)
{
    int mt  = row >> 4;
    int kt  = k >> 5;
    int kin = k & 31;
    int g, j;
    if (kin < 16) { g = kin >> 3;        j = kin & 7; }
    else          { g = (kin - 16) >> 3; j = 8 + ((kin - 16) & 7); }
    int lane = (g << 4) | (row & 15);
    return (((size_t)mt * apStride + kt0 + kt) * 32 + lane) * 16 + j;
}

// ---------------------------------------------------------------------------
// Weight packing: fp32 KxN row-major -> per-lane bf16 WMMA B-fragments.
//   lane = 16*((kin>>3)&1) + (n%16), j = (kin&7) + 8*(kin>>4)  (table-order K per lane)
// ---------------------------------------------------------------------------
__global__ void pack_w_kernel(const float* __restrict__ W, int srcK, int srcN,
                              int Kblk, int Nblk, int koff, int noff,
                              int Kb, unsigned short* __restrict__ dst)
{
    long long idx = (long long)blockIdx.x * blockDim.x + threadIdx.x;
    long long tot = (long long)Kblk * Nblk;
    if (idx >= tot) return;
    int ks = (int)(idx / Nblk);
    int ns = (int)(idx % Nblk);
    float v = (ks < srcK && ns < srcN) ? W[(size_t)ks * srcN + ns] : 0.0f;
    int k = koff + ks, n = noff + ns;
    int kt = k >> 5, kin = k & 31;
    int nt = n >> 4, nin = n & 15;
    int lane = (((kin >> 3) & 1) << 4) | nin;
    int j = (kin & 7) + ((kin >> 4) << 3);
    int Kt = Kb >> 5;
    dst[(((size_t)nt * Kt + kt) * 32 + lane) * 16 + j] = f2bf(v);
}

// fp32 MxK (ldX) -> packed bf16 A fragments (apStride = Kt, kt0 = 0)
__global__ void pack_a_kernel(const float* __restrict__ X, int ldX, int M, int K,
                              unsigned short* __restrict__ dst)
{
    int Kt = K >> 5;
    long long tot = (long long)((M + 15) >> 4) * Kt * 512;
    long long idx = (long long)blockIdx.x * blockDim.x + threadIdx.x;
    if (idx >= tot) return;
    int j    = (int)(idx & 15);
    int lane = (int)((idx >> 4) & 31);
    long long tile = idx >> 9;
    int kt = (int)(tile % Kt);
    int mt = (int)(tile / Kt);
    int g  = lane >> 4;
    int row = (mt << 4) + (lane & 15);
    int k = (kt << 5) + (g << 3) + ((j < 8) ? j : (j + 8));
    float v = (row < M) ? X[(size_t)row * ldX + k] : 0.f;
    dst[idx] = f2bf(v);
}

// ---------------------------------------------------------------------------
// Generic WMMA GEMM with packed bf16 A and W operands.
//   Y[M,N](fp32, optional) = act( A * W + bias ),
//   plus optional packed-bf16 output (A-fragment layout for the next GEMM).
// One wave per 16x16 tile, 4 waves/block. act: 0=none,1=relu,2=tanh
// ---------------------------------------------------------------------------
__global__ void gemm_wmma_kernel(const unsigned short* __restrict__ Ap, int apStride, int apKt0,
                                 const unsigned short* __restrict__ Wp,
                                 const float* __restrict__ bias,
                                 float* __restrict__ Y, int ldY,
                                 unsigned short* __restrict__ Yp, int ypStride, int ypKt0,
                                 int M, int N, int K, int act, int Nvalid)
{
    const int lane = threadIdx.x & 31;
    const int Nt = N >> 4;
    const int Mt = (M + 15) >> 4;
    int wave = blockIdx.x * (blockDim.x >> 5) + (threadIdx.x >> 5);
    if (wave >= Mt * Nt) return;                 // wave-uniform: EXEC stays all-ones
    const int nt = wave % Nt;
    const int mt = wave / Nt;
    const int Kt = K >> 5;
    const int g  = lane >> 4;
    const int lm = lane & 15;
    const int m0 = mt << 4;

    const unsigned short* ap = Ap + (((size_t)mt * apStride + apKt0) * 32 + lane) * 16;
    const unsigned short* wp = Wp + ((size_t)nt * Kt * 32 + lane) * 16;

    f32x8 acc0 = {0.f,0.f,0.f,0.f,0.f,0.f,0.f,0.f};
    f32x8 acc1 = {0.f,0.f,0.f,0.f,0.f,0.f,0.f,0.f};
    int kt = 0;
    for (; kt + 1 < Kt; kt += 2) {
        u16x16 a0 = *(const u16x16*)(ap + (size_t)kt * 512);
        u16x16 b0 = *(const u16x16*)(wp + (size_t)kt * 512);
        u16x16 a1 = *(const u16x16*)(ap + (size_t)(kt + 1) * 512);
        u16x16 b1 = *(const u16x16*)(wp + (size_t)(kt + 1) * 512);
        acc0 = __builtin_amdgcn_wmma_f32_16x16x32_bf16(
                   false, __builtin_bit_cast(bf16x16, a0), false,
                   __builtin_bit_cast(bf16x16, b0), (short)0, acc0, false, false);
        acc1 = __builtin_amdgcn_wmma_f32_16x16x32_bf16(
                   false, __builtin_bit_cast(bf16x16, a1), false,
                   __builtin_bit_cast(bf16x16, b1), (short)0, acc1, false, false);
    }
    if (kt < Kt) {
        u16x16 a0 = *(const u16x16*)(ap + (size_t)kt * 512);
        u16x16 b0 = *(const u16x16*)(wp + (size_t)kt * 512);
        acc0 = __builtin_amdgcn_wmma_f32_16x16x32_bf16(
                   false, __builtin_bit_cast(bf16x16, a0), false,
                   __builtin_bit_cast(bf16x16, b0), (short)0, acc0, false, false);
    }

    // D layout: VGPR r -> M = m0 + r + 8*g, N = nt*16 + (lane&15)
    const int n = (nt << 4) + lm;
    float bv = 0.f;
    if (bias && n < Nvalid) bv = bias[n];
#pragma unroll
    for (int r = 0; r < 8; ++r) {
        int m = m0 + r + (g << 3);
        if (m < M && n < Nvalid) {
            float v = acc0[r] + acc1[r] + bv;
            if (act == 1)      v = fmaxf(v, 0.f);
            else if (act == 2) v = tanhf(v);
            if (Y)  Y[(size_t)m * ldY + n] = v;
            if (Yp) Yp[packA_idx(m, n, ypStride, ypKt0)] = f2bf(v);
        }
    }
}

// ---------------------------------------------------------------------------
// Small helper kernels
// ---------------------------------------------------------------------------
__global__ void bias_combine_kernel(const float* bih, const float* bhh,
                                    const float* bi2h, const float* bh2h, float* b1)
{
    int n = blockIdx.x * blockDim.x + threadIdx.x;
    if (n >= GN) return;
    b1[n] = (n < 2048) ? (bih[n] + bhh[n]) : (bi2h[n - 2048] + bh2h[n - 2048]);
}

// zero Az tiles 8..39 (prev_out + h) for both m-tiles, and c state
__global__ void init_state_kernel(unsigned short* Az, float* c)
{
    int idx = blockIdx.x * blockDim.x + threadIdx.x;
    if (idx < 2 * 32 * 512) {
        int mt = idx >> 14;
        int r  = idx & 16383;
        Az[((size_t)mt * ZKT + 8) * 512 + r] = 0;       // tiles 8..39 = 32*512 ushorts
    }
    if (idx < BB * HID) c[idx] = 0.f;
}

// xt = embed[seq[:,t]]  -> packed into Az tiles 0..7
__global__ void embed_gather_kernel(const int* __restrict__ seq,
                                    const float* __restrict__ embed,
                                    unsigned short* __restrict__ Az, int t)
{
    int idx = blockIdx.x * blockDim.x + threadIdx.x;    // BB*EMB
    if (idx >= BB * EMB) return;
    int b = idx >> 8, j = idx & 255;
    int tok = seq[b * TT + t];
    Az[packA_idx(b, j, ZKT, 0)] = f2bf(embed[(size_t)tok * EMB + j]);
}

// LSTM pointwise: h_n -> Az tiles 24..39 (packed); fr_pre -> packed A buffer
__global__ void lstm_pointwise_kernel(const float* __restrict__ G,
                                      float* __restrict__ c,
                                      unsigned short* __restrict__ Az,
                                      unsigned short* __restrict__ Afrp)
{
    int idx = blockIdx.x * blockDim.x + threadIdx.x;    // BB*HID
    if (idx >= BB * HID) return;
    int b = idx >> 9, j = idx & 511;
    const float* Gb = G + (size_t)b * GN;
    float iv = sigf(Gb[j]);
    float fv = sigf(Gb[512 + j]);
    float gv = tanhf(Gb[1024 + j]);
    float ov = sigf(Gb[1536 + j]);
    float n5 = Gb[2048 + j];
    float cn = fv * c[idx] + iv * gv;
    float tc = tanhf(cn);
    c[idx] = cn;
    Az[packA_idx(b, 768 + j, ZKT, 0)] = f2bf(ov * tc);  // h_n
    Afrp[packA_idx(b, j, 16, 0)]      = f2bf(sigf(n5) * tc);
}

// Fused attention: scores(257) -> softmax -> context; writes (vis+hol) packed.
__global__ void attention_kernel(const float* __restrict__ ctx_embed,
                                 const float* __restrict__ cnn,
                                 const float* __restrict__ fr,
                                 const float* __restrict__ fre,
                                 const float* __restrict__ hoe,
                                 const float* __restrict__ hol,
                                 const float* __restrict__ Wa,
                                 const float* __restrict__ ba,
                                 unsigned short* __restrict__ Aatt)
{
    int b = blockIdx.x, tid = threadIdx.x;              // blockDim = 512
    __shared__ float shh[HID], swa[HID], sc[ATT + 1];
    __shared__ float s_mx, s_sum;
    shh[tid] = hoe[(size_t)b * HID + tid];
    swa[tid] = Wa[tid];
    __syncthreads();

    if (tid < ATT + 1) {
        const float* src = (tid == 0) ? (fre + (size_t)b * HID)
                                      : (ctx_embed + ((size_t)b * ATT + (tid - 1)) * HID);
        float s = 0.f;
        for (int j = 0; j < HID; ++j) s += tanhf(src[j] + shh[j]) * swa[j];
        sc[tid] = s + ba[0];
    }
    __syncthreads();
    if (tid == 0) {
        float m = -3.4e38f;
        for (int i = 0; i < ATT + 1; ++i) m = fmaxf(m, sc[i]);
        float su = 0.f;
        for (int i = 0; i < ATT + 1; ++i) su += expf(sc[i] - m);
        s_mx = m; s_sum = su;
    }
    __syncthreads();
    float inv = 1.f / s_sum;
    for (int i = tid; i < ATT + 1; i += 512) sc[i] = expf(sc[i] - s_mx) * inv;
    __syncthreads();

    float acc = sc[0] * fr[(size_t)b * CTX + tid];      // sentinel term
    const float* cb = cnn + (size_t)b * ATT * CTX + tid;
    for (int a = 0; a < ATT; ++a) acc += sc[a + 1] * cb[(size_t)a * CTX];
    acc += hol[(size_t)b * HID + tid];                  // vis + hol
    Aatt[packA_idx(b, tid, 16, 0)] = f2bf(acc);
}

__global__ void logsoftmax_kernel(const float* __restrict__ logits,
                                  float* __restrict__ out, int t)
{
    int b = blockIdx.x, tid = threadIdx.x;              // blockDim = 256
    __shared__ float sbuf[VOC];
    __shared__ float red[256];
    const float* lb = logits + (size_t)b * VOC;
    for (int i = tid; i < VOC; i += 256) sbuf[i] = lb[i];
    __syncthreads();
    float m = -3.4e38f;
    for (int i = tid; i < VOC; i += 256) m = fmaxf(m, sbuf[i]);
    red[tid] = m; __syncthreads();
    for (int s = 128; s > 0; s >>= 1) { if (tid < s) red[tid] = fmaxf(red[tid], red[tid + s]); __syncthreads(); }
    float mx = red[0]; __syncthreads();
    float ss = 0.f;
    for (int i = tid; i < VOC; i += 256) ss += expf(sbuf[i] - mx);
    red[tid] = ss; __syncthreads();
    for (int s = 128; s > 0; s >>= 1) { if (tid < s) red[tid] += red[tid + s]; __syncthreads(); }
    float ls = logf(red[0]) + mx;
    float* ob = out + ((size_t)b * NSTEP + t) * VOC;
    for (int i = tid; i < VOC; i += 256) ob[i] = sbuf[i] - ls;
}

// ---------------------------------------------------------------------------
// Host side
// ---------------------------------------------------------------------------
static void launch_gemm(const unsigned short* Ap, int apStride, int apKt0,
                        const unsigned short* Wp, const float* bias,
                        float* Y, int ldY,
                        unsigned short* Yp, int ypStride, int ypKt0,
                        int M, int N, int K, int act, int Nvalid, hipStream_t s)
{
    int tiles = ((M + 15) / 16) * (N / 16);
    int blocks = (tiles + 3) / 4;
    gemm_wmma_kernel<<<blocks, 128, 0, s>>>(Ap, apStride, apKt0, Wp, bias, Y, ldY,
                                            Yp, ypStride, ypKt0, M, N, K, act, Nvalid);
}

static void launch_pack(const float* W, int srcK, int srcN, int Kblk, int Nblk,
                        int koff, int noff, int Kb, unsigned short* dst, hipStream_t s)
{
    long long tot = (long long)Kblk * Nblk;
    int blocks = (int)((tot + 255) / 256);
    pack_w_kernel<<<blocks, 256, 0, s>>>(W, srcK, srcN, Kblk, Nblk, koff, noff, Kb, dst);
}

extern "C" void kernel_launch(void* const* d_in, const int* in_sizes, int n_in,
                              void* d_out, int out_size, void* d_ws, size_t ws_size,
                              hipStream_t stream)
{
    (void)in_sizes; (void)n_in; (void)out_size; (void)ws_size;
    const float* cnn   = (const float*)d_in[0];
    const int*   seq   = (const int*)d_in[1];
    const float* embed = (const float*)d_in[2];
    const float* Wce   = (const float*)d_in[3];
    const float* bce   = (const float*)d_in[4];
    const float* Wih   = (const float*)d_in[5];
    const float* bih   = (const float*)d_in[6];
    const float* Whh   = (const float*)d_in[7];
    const float* bhh   = (const float*)d_in[8];
    const float* Wi2h  = (const float*)d_in[9];
    const float* bi2h  = (const float*)d_in[10];
    const float* Wh2h  = (const float*)d_in[11];
    const float* bh2h  = (const float*)d_in[12];
    const float* Wfr   = (const float*)d_in[13];
    const float* bfr   = (const float*)d_in[14];
    const float* Wfre  = (const float*)d_in[15];
    const float* bfre  = (const float*)d_in[16];
    const float* Who   = (const float*)d_in[17];
    const float* bho   = (const float*)d_in[18];
    const float* Whoe  = (const float*)d_in[19];
    const float* bhoe  = (const float*)d_in[20];
    const float* Wa    = (const float*)d_in[21];
    const float* ba    = (const float*)d_in[22];
    const float* Watt  = (const float*)d_in[23];
    const float* batt  = (const float*)d_in[24];
    const float* Wlog  = (const float*)d_in[25];
    const float* blog  = (const float*)d_in[26];
    float* out = (float*)d_out;

    // workspace carve-up (all offsets 256B aligned)
    char* base = (char*)d_ws;
    size_t off = 0;
    auto alloc = [&](size_t bytes) -> char* {
        char* p = base + off;
        off = (off + bytes + 255) & ~(size_t)255;
        return p;
    };
    const int NLOGP = 5008;                              // VOCAB padded to /16
    unsigned short* W1p   = (unsigned short*)alloc((size_t)ZK * GN * 2);
    unsigned short* Wcep  = (unsigned short*)alloc((size_t)CTX * HID * 2);
    unsigned short* Wfrp  = (unsigned short*)alloc((size_t)HID * CTX * 2);
    unsigned short* Wfrep = (unsigned short*)alloc((size_t)CTX * HID * 2);
    unsigned short* Whop  = (unsigned short*)alloc((size_t)HID * HID * 2);
    unsigned short* Whoep = (unsigned short*)alloc((size_t)HID * HID * 2);
    unsigned short* Wattp = (unsigned short*)alloc((size_t)HID * HID * 2);
    unsigned short* Wlogp = (unsigned short*)alloc((size_t)HID * NLOGP * 2);
    unsigned short* Acnn  = (unsigned short*)alloc((size_t)(BB * ATT / 16) * 16 * 512 * 2);
    unsigned short* Az    = (unsigned short*)alloc((size_t)2 * ZKT * 512 * 2);
    unsigned short* Afrp  = (unsigned short*)alloc((size_t)2 * 16 * 512 * 2);  // fr_pre
    unsigned short* Afr2  = (unsigned short*)alloc((size_t)2 * 16 * 512 * 2);  // fr
    unsigned short* Ahol  = (unsigned short*)alloc((size_t)2 * 16 * 512 * 2);  // hol
    unsigned short* Aatt  = (unsigned short*)alloc((size_t)2 * 16 * 512 * 2);  // vis+hol
    float* b1     = (float*)alloc(GN * 4);
    float* ctxe   = (float*)alloc((size_t)BB * ATT * HID * 4);
    float* G      = (float*)alloc((size_t)BB * GN * 4);
    float* cbuf   = (float*)alloc((size_t)BB * HID * 4);
    float* frb    = (float*)alloc((size_t)BB * CTX * 4);
    float* freb   = (float*)alloc((size_t)BB * HID * 4);
    float* holb   = (float*)alloc((size_t)BB * HID * 4);
    float* hoeb   = (float*)alloc((size_t)BB * HID * 4);
    float* logits = (float*)alloc((size_t)BB * VOC * 4);

    // ---- one-time setup (re-done every call; deterministic) ----
    // Fused gate/n5 weight: [ Wih ; Whh | Wi2h ; Wh2h ]  (1280 x 2560)
    launch_pack(Wih,  768, 2048, 768, 2048,   0,    0, ZK, W1p, stream);
    launch_pack(Whh,  512, 2048, 512, 2048, 768,    0, ZK, W1p, stream);
    launch_pack(Wi2h, 768,  512, 768,  512,   0, 2048, ZK, W1p, stream);
    launch_pack(Wh2h, 512,  512, 512,  512, 768, 2048, ZK, W1p, stream);
    launch_pack(Wce,  512,  512, 512,  512, 0, 0, CTX, Wcep,  stream);
    launch_pack(Wfr,  512,  512, 512,  512, 0, 0, HID, Wfrp,  stream);
    launch_pack(Wfre, 512,  512, 512,  512, 0, 0, CTX, Wfrep, stream);
    launch_pack(Who,  512,  512, 512,  512, 0, 0, HID, Whop,  stream);
    launch_pack(Whoe, 512,  512, 512,  512, 0, 0, HID, Whoep, stream);
    launch_pack(Watt, 512,  512, 512,  512, 0, 0, HID, Wattp, stream);
    launch_pack(Wlog, 512, VOC,  512, NLOGP, 0, 0, HID, Wlogp, stream);

    {   // pack cnn_feats (8192 x 512) as A fragments for the ctx_embed GEMM
        long long tot = (long long)(BB * ATT / 16) * 16 * 512;
        pack_a_kernel<<<(int)((tot + 255) / 256), 256, 0, stream>>>(cnn, CTX, BB * ATT, CTX, Acnn);
    }

    bias_combine_kernel<<<(GN + 255) / 256, 256, 0, stream>>>(bih, bhh, bi2h, bh2h, b1);
    init_state_kernel<<<(2 * 32 * 512 + 255) / 256, 256, 0, stream>>>(Az, cbuf);

    // ctx_embed = relu(cnn_feats @ Wce + bce)   (8192 x 512, K=512)
    launch_gemm(Acnn, 16, 0, Wcep, bce, ctxe, HID, nullptr, 0, 0,
                BB * ATT, HID, CTX, 1, HID, stream);

    // ---- recurrent scan ----
    for (int t = 0; t < NSTEP; ++t) {
        embed_gather_kernel<<<(BB * EMB + 255) / 256, 256, 0, stream>>>(seq, embed, Az, t);

        // gates ++ n5 :  (32 x 2560) = z(32x1280) @ W1 + b1
        launch_gemm(Az, ZKT, 0, W1p, b1, G, GN, nullptr, 0, 0,
                    BB, GN, ZK, 0, GN, stream);

        lstm_pointwise_kernel<<<(BB * HID + 255) / 256, 256, 0, stream>>>(G, cbuf, Az, Afrp);

        // fr = relu(fr_pre @ Wfr + bfr)           (also packed for fre GEMM)
        launch_gemm(Afrp, 16, 0, Wfrp, bfr, frb, CTX, Afr2, 16, 0,
                    BB, CTX, HID, 1, CTX, stream);
        // hol = tanh(h_n @ Who + bho); h_n read from Az tiles 24..39
        launch_gemm(Az, ZKT, 24, Whop, bho, holb, HID, Ahol, 16, 0,
                    BB, HID, HID, 2, HID, stream);
        // fre = fr @ Wfre + bfre ; hoe = hol @ Whoe + bhoe
        launch_gemm(Afr2, 16, 0, Wfrep, bfre, freb, HID, nullptr, 0, 0,
                    BB, HID, CTX, 0, HID, stream);
        launch_gemm(Ahol, 16, 0, Whoep, bhoe, hoeb, HID, nullptr, 0, 0,
                    BB, HID, HID, 0, HID, stream);

        attention_kernel<<<BB, 512, 0, stream>>>(ctxe, cnn, frb, freb, hoeb, holb, Wa, ba, Aatt);

        // out_h = tanh((vis+hol) @ Watt + batt) -> packed into Az tiles 8..23
        launch_gemm(Aatt, 16, 0, Wattp, batt, nullptr, 0, Az, ZKT, 8,
                    BB, HID, HID, 2, HID, stream);

        // logits = out_h @ Wlog + blog ; out_h read from Az tiles 8..23
        launch_gemm(Az, ZKT, 8, Wlogp, blog, logits, VOC, nullptr, 0, 0,
                    BB, NLOGP, HID, 0, VOC, stream);

        logsoftmax_kernel<<<BB, 256, 0, stream>>>(logits, out, t);
    }
}
